// RCAttention_74500502716587
// MI455X (gfx1250) — matrix operand order repaired and checked by reading, not compiled, not run on hardware
//
#include <hip/hip_runtime.h>
#include <stdint.h>

typedef __attribute__((ext_vector_type(16))) __bf16 v16bf;
typedef __attribute__((ext_vector_type(8)))  float  v8f;

#define BLOCK  256
#define M_TILE 128
#define N_TILE 64
#define K_TILE 32
#define PITCH  40   // bf16 elements per LDS row (K_TILE + 8 pad); 80B rows keep 16B alignment

union Frag {
    uint4 q[2];
    v16bf v;
};
static_assert(sizeof(Frag) == 32, "frag size");

__device__ __forceinline__ unsigned short f2bf(float f) {
    unsigned u = __float_as_uint(f);
    u += 0x7FFFu + ((u >> 16) & 1u);   // round-to-nearest-even
    return (unsigned short)(u >> 16);
}

// C[m,n] = sum_k A[m,k] * B[.,.]  (+bias[n]) (+existing C)
// A: bf16, row-major [m][k], row stride lda (elements)
// B: bf16. BT=false: stored [n][k], row stride ldb.  BT=true: stored [k][n], row stride ldb.
// Per-z slice offsets: off = (z>>8)*s1 + (z&255)*s2  (elements) for A, B, C independently.
template<bool BT, bool BIAS, bool ACCUM, bool OUT_BF16>
__global__ __launch_bounds__(BLOCK)
void gemm_bf16_kernel(const unsigned short* __restrict__ A,
                      const unsigned short* __restrict__ Bm,
                      void* __restrict__ Cout,
                      const float* __restrict__ bias,
                      int K, int lda, int ldb, int ldc,
                      long sA1, long sA2, long sB1, long sB2,
                      long sC1, long sC2)
{
    // double-buffered LDS tiles: one barrier per K-step
    __shared__ __align__(16) unsigned short lsA[2][M_TILE * PITCH];
    __shared__ __align__(16) unsigned short lsB[2][N_TILE * PITCH];

    const int t    = threadIdx.x;
    const int lane = t & 31;
    const int wid  = t >> 5;
    const int wm   = wid & 3;     // wave row   (4 waves along M)
    const int wn   = wid >> 2;    // wave col   (2 waves along N)
    const int l16  = lane & 15;
    const int half = lane >> 4;

    const long z  = blockIdx.z;
    const long zb = z >> 8, zr = z & 255;
    const unsigned short* Ab = A  + zb * sA1 + zr * sA2;
    const unsigned short* Bb = Bm + zb * sB1 + zr * sB2;
    const long coff          =      zb * sC1 + zr * sC2;

    const int m0 = blockIdx.y * M_TILE;
    const int n0 = blockIdx.x * N_TILE;

    // ---- staging helpers (b128 global loads) ----
    uint4 ra[2];            // A tile: 128x32 bf16 = 512 x 16B chunks, 2/thread
    uint4 rb;               // B tile (!BT): 64x32 bf16 = 256 chunks, 1/thread
    unsigned rbt[4];        // B tile (BT): b32 loads, 2 bf16 along n, 4/thread

    auto loadA = [&](int k0) {
        #pragma unroll
        for (int i = 0; i < 2; ++i) {
            int idx = t + i * BLOCK;          // 0..511
            int r   = idx >> 2;               // 0..127
            int kc  = (idx & 3) * 8;          // 0,8,16,24
            ra[i] = *reinterpret_cast<const uint4*>(Ab + (long)(m0 + r) * lda + k0 + kc);
        }
    };
    auto storeA = [&](int buf) {
        #pragma unroll
        for (int i = 0; i < 2; ++i) {
            int idx = t + i * BLOCK;
            int r   = idx >> 2;
            int kc  = (idx & 3) * 8;
            *reinterpret_cast<uint4*>(&lsA[buf][r * PITCH + kc]) = ra[i];
        }
    };
    auto loadB = [&](int k0) {
        if (!BT) {
            int r  = t >> 2;                  // 0..63
            int kc = (t & 3) * 8;
            rb = *reinterpret_cast<const uint4*>(Bb + (long)(n0 + r) * ldb + k0 + kc);
        } else {
            #pragma unroll
            for (int i = 0; i < 4; ++i) {
                int e  = t + i * BLOCK;       // 0..1023
                int n2 = (e & 31) * 2;        // 0..62, coalesced along n
                int kk = e >> 5;              // 0..31
                rbt[i] = *reinterpret_cast<const unsigned*>(Bb + (long)(k0 + kk) * ldb + n0 + n2);
            }
        }
    };
    auto storeB = [&](int buf) {
        if (!BT) {
            int r  = t >> 2;
            int kc = (t & 3) * 8;
            *reinterpret_cast<uint4*>(&lsB[buf][r * PITCH + kc]) = rb;
        } else {
            #pragma unroll
            for (int i = 0; i < 4; ++i) {
                int e  = t + i * BLOCK;
                int n2 = (e & 31) * 2;
                int kk = e >> 5;
                lsB[buf][n2 * PITCH + kk]       = (unsigned short)(rbt[i] & 0xFFFFu);
                lsB[buf][(n2 + 1) * PITCH + kk] = (unsigned short)(rbt[i] >> 16);
            }
        }
    };

    v8f acc[2][2] = {};

    // prologue: stage tile 0
    loadA(0);
    loadB(0);
    storeA(0);
    storeB(0);
    __syncthreads();

    const int nsteps = K / K_TILE;
    for (int s = 0; s < nsteps; ++s) {
        const int cur = s & 1;
        const int nxt = cur ^ 1;

        if (s + 1 < nsteps) {                 // issue next-tile global loads early
            loadA((s + 1) * K_TILE);
            loadB((s + 1) * K_TILE);
        }

        // ---- fragments from current buffer ----
        Frag a[2], b[2];
        #pragma unroll
        for (int mi = 0; mi < 2; ++mi) {
            int r = wm * 32 + mi * 16 + l16;                 // A: lane%16 -> M
            const unsigned short* base = &lsA[cur][r * PITCH];
            a[mi].q[0] = *reinterpret_cast<const uint4*>(base + half * 8);        // K 0..7 / 8..15
            a[mi].q[1] = *reinterpret_cast<const uint4*>(base + 16 + half * 8);   // K 16..23 / 24..31
        }
        #pragma unroll
        for (int ni = 0; ni < 2; ++ni) {
            int r = wn * 32 + ni * 16 + l16;                 // B: lane%16 -> N
            const unsigned short* base = &lsB[cur][r * PITCH];                    // K 0..7,16..23 / 8..15,24..31
            b[ni].q[0] = *reinterpret_cast<const uint4*>(base + half * 8);
            b[ni].q[1] = *reinterpret_cast<const uint4*>(base + 16 + half * 8);
        }
        #pragma unroll
        for (int mi = 0; mi < 2; ++mi)
            #pragma unroll
            for (int ni = 0; ni < 2; ++ni)
                acc[mi][ni] = __builtin_amdgcn_wmma_f32_16x16x32_bf16(
                    false, a[mi].v, false, b[ni].v, (short)0, acc[mi][ni], false, false);

        if (s + 1 < nsteps) {                 // fill the other buffer while this one is read
            storeA(nxt);
            storeB(nxt);
        }
        __syncthreads();
    }

    // ---- epilogue: C/D layout: lane%16 -> N, (lane/16)*8 + vgpr -> M ----
    #pragma unroll
    for (int mi = 0; mi < 2; ++mi) {
        #pragma unroll
        for (int ni = 0; ni < 2; ++ni) {
            int col  = n0 + wn * 32 + ni * 16 + l16;
            float bv = BIAS ? bias[col] : 0.0f;
            #pragma unroll
            for (int r = 0; r < 8; ++r) {
                int row  = m0 + wm * 32 + mi * 16 + half * 8 + r;
                long off = coff + (long)row * ldc + col;
                float val = acc[mi][ni][r] + bv;
                if (OUT_BF16) {
                    ((unsigned short*)Cout)[off] = f2bf(val);
                } else {
                    float* Cf = (float*)Cout;
                    if (ACCUM) val += Cf[off];
                    Cf[off] = val;
                }
            }
        }
    }
}

__global__ __launch_bounds__(BLOCK)
void f32_to_bf16_kernel(const float* __restrict__ in, unsigned short* __restrict__ out, long n4) {
    long i      = (long)blockIdx.x * blockDim.x + threadIdx.x;
    long stride = (long)gridDim.x * blockDim.x;
    for (; i < n4; i += stride) {
        float4 f = reinterpret_cast<const float4*>(in)[i];
        uint2 o;
        o.x = (unsigned)f2bf(f.x) | ((unsigned)f2bf(f.y) << 16);
        o.y = (unsigned)f2bf(f.z) | ((unsigned)f2bf(f.w) << 16);
        reinterpret_cast<uint2*>(out)[i] = o;
    }
}

// one wave32 per 256-wide row; 8 rows per block
__global__ __launch_bounds__(BLOCK)
void softmax_kernel(const float* __restrict__ S, unsigned short* __restrict__ P) {
    const int lane = threadIdx.x & 31;
    const int wid  = threadIdx.x >> 5;
    const long row = (long)blockIdx.x * 8 + wid;
    const float* src = S + row * 256;
    float v[8];
    float mx = -3.4e38f;
    #pragma unroll
    for (int i = 0; i < 8; ++i) { v[i] = src[lane + i * 32]; mx = fmaxf(mx, v[i]); }
    #pragma unroll
    for (int s = 16; s >= 1; s >>= 1) mx = fmaxf(mx, __shfl_xor(mx, s, 32));
    float sum = 0.f;
    #pragma unroll
    for (int i = 0; i < 8; ++i) { v[i] = __expf(v[i] - mx); sum += v[i]; }
    #pragma unroll
    for (int s = 16; s >= 1; s >>= 1) sum += __shfl_xor(sum, s, 32);
    float inv = 1.0f / sum;
    unsigned short* dst = P + row * 256;
    #pragma unroll
    for (int i = 0; i < 8; ++i) dst[lane + i * 32] = f2bf(v[i] * inv);
}

extern "C" void kernel_launch(void* const* d_in, const int* in_sizes, int n_in,
                              void* d_out, int out_size, void* d_ws, size_t ws_size,
                              hipStream_t stream) {
    constexpr int  Cdim = 256, Bdim = 2, H = 256, W = 256;
    constexpr long N  = (long)Cdim * Cdim;   // 65536
    constexpr long MN = (long)Bdim * N;      // 131072 rows of x

    const float* x  = (const float*)d_in[0];
    const float* Wq = (const float*)d_in[1];
    const float* bq = (const float*)d_in[2];
    const float* Wk = (const float*)d_in[3];
    const float* bk = (const float*)d_in[4];
    const float* Wv = (const float*)d_in[5];
    const float* bv = (const float*)d_in[6];

    char* ws = (char*)d_ws;
    unsigned short* xb  = (unsigned short*)(ws);                 // 64 MB (reused as P later)
    unsigned short* qb  = (unsigned short*)(ws + (1L << 26));    // 64 MB
    unsigned short* kb  = (unsigned short*)(ws + (2L << 26));    // 64 MB
    unsigned short* vb  = (unsigned short*)(ws + (3L << 26));    // 64 MB
    float*          S   = (float*)         (ws + (4L << 26));    // 128 MB  (B*H*W*W f32)
    unsigned short* wqb = (unsigned short*)(ws + (6L << 26));
    unsigned short* wkb = wqb + 65536;
    unsigned short* wvb = wkb + 65536;
    unsigned short* P   = xb;   // x no longer needed after projections

    // 1) convert inputs to bf16
    f32_to_bf16_kernel<<<4096, BLOCK, 0, stream>>>(x,  xb,  MN * Cdim / 4);
    f32_to_bf16_kernel<<<64,   BLOCK, 0, stream>>>(Wq, wqb, 65536 / 4);
    f32_to_bf16_kernel<<<64,   BLOCK, 0, stream>>>(Wk, wkb, 65536 / 4);
    f32_to_bf16_kernel<<<64,   BLOCK, 0, stream>>>(Wv, wvb, 65536 / 4);

    // 2) q/k/v = x @ W^T + b      (W stored [o][c] == [n][k] -> BT=false; bf16 out)
    dim3 gp(Cdim / N_TILE, MN / M_TILE, 1);   // (4, 1024, 1)
    gemm_bf16_kernel<false, true, false, true><<<gp, BLOCK, 0, stream>>>(
        xb, wqb, qb, bq, Cdim, Cdim, Cdim, Cdim, 0, 0, 0, 0, 0, 0);
    gemm_bf16_kernel<false, true, false, true><<<gp, BLOCK, 0, stream>>>(
        xb, wkb, kb, bk, Cdim, Cdim, Cdim, Cdim, 0, 0, 0, 0, 0, 0);
    gemm_bf16_kernel<false, true, false, true><<<gp, BLOCK, 0, stream>>>(
        xb, wvb, vb, bv, Cdim, Cdim, Cdim, Cdim, 0, 0, 0, 0, 0, 0);

    // 3) row logits: per (b,h): S = q[b,h] @ k[b,h]   (k slice is [c][o] -> BT=true)
    dim3 ga(W / N_TILE, W / M_TILE, Bdim * H); // (4, 2, 512)
    gemm_bf16_kernel<true, false, false, false><<<ga, BLOCK, 0, stream>>>(
        qb, kb, S, nullptr, Cdim, Cdim, Cdim, W,
        16777216L, 65536L, 16777216L, 65536L, 16777216L, 65536L);

    // 4) col logits: per (b,w): S += q[b,:,w,:] @ k[b,:,w,:]^T  ([G][c] -> BT=false, big strides)
    gemm_bf16_kernel<false, false, true, false><<<ga, BLOCK, 0, stream>>>(
        qb, kb, S, nullptr, Cdim, W * Cdim, W * Cdim, W * W,
        16777216L, 256L, 16777216L, 256L, 16777216L, 256L);

    // 5) softmax over last axis -> bf16 P
    softmax_kernel<<<(unsigned)(Bdim * H * W / 8), BLOCK, 0, stream>>>(S, P);

    // 6) out = P @ v   (v slice is [o][c] -> BT=true), f32 out to d_out
    gemm_bf16_kernel<true, false, false, false><<<ga, BLOCK, 0, stream>>>(
        P, vb, d_out, nullptr, W, W, Cdim, Cdim,
        16777216L, 65536L, 16777216L, 65536L, 16777216L, 65536L);
}
